// BlockGRU_27436251087184
// MI455X (gfx1250) — compile-verified, matrix operand order checked by reading, not run on hardware
//
#include <hip/hip_runtime.h>
#include <hip/hip_bf16.h>
#include <stdint.h>
#include <stddef.h>

// Problem sizes (fixed by the reference)
#define Hh    256      // hidden size
#define BATCH 64
#define SEQ   2048
#define NWG   16       // persistent workgroups; each owns 16 of 256 hidden cols
#define TPB   128      // 4 waves (wave32); wave w owns M-tile w
#define STEP_DW 8192   // dwords in one 64x256 bf16 fragment-ordered tile (32 KB)

typedef __attribute__((ext_vector_type(16))) __bf16 v16bf;
typedef __attribute__((ext_vector_type(8)))  float  v8f;
typedef int v4i __attribute__((vector_size(16)));
typedef __attribute__((address_space(1))) v4i* g_v4i_ptr;
typedef __attribute__((address_space(3))) v4i* l_v4i_ptr;

#if defined(__HIP_DEVICE_COMPILE__) && defined(__has_builtin)
#if __has_builtin(__builtin_amdgcn_global_load_async_to_lds_b128)
#define HAVE_ASYNC 1
#endif
#endif
#ifndef HAVE_ASYNC
#define HAVE_ASYNC 0
#endif

__device__ __forceinline__ uint16_t f2bf(float f) {
  uint32_t u = __builtin_bit_cast(uint32_t, f);
  u += 0x7FFFu + ((u >> 16) & 1u);            // round-to-nearest-even
  return (uint16_t)(u >> 16);
}
__device__ __forceinline__ float bf2f(uint16_t h) {
  uint32_t u = ((uint32_t)h) << 16;
  return __builtin_bit_cast(float, u);
}

// u16 index of element (m, k) inside a fragment-ordered 64x256 bf16 tile laid
// out as [mt][kt][lane][v] dwords (WMMA 16-bit A/B layout, ISA 7.12.2).
__device__ __forceinline__ int sw_u16(int m, int k) {
  int mtt = m >> 4, kt = k >> 5;
  int rem = k & 31;
  int g2  = rem >> 4;          // selects VGPR quad
  int hi  = (rem >> 3) & 1;    // lane half
  int sub = rem & 7;
  int v   = g2 * 4 + (sub >> 1);
  int half = sub & 1;
  int ln  = hi * 16 + (m & 15);
  return (((((mtt * 8 + kt) * 32) + ln) * 8 + v) << 1) | half;
}

union Frag { uint32_t u[8]; v16bf v; };

__device__ __forceinline__ v16bf load_frag(const uint32_t* p) {
  Frag f;
#pragma unroll
  for (int i = 0; i < 8; ++i) f.u[i] = p[i];   // -> ds_load_b128 x2
  return f.v;
}

// Contiguous 32 KB global -> LDS stage (fragment-ordered on both sides).
__device__ __forceinline__ void stage_issue(const uint32_t* g, uint32_t* l) {
#if HAVE_ASYNC
  for (int c = threadIdx.x; c < STEP_DW / 4; c += TPB)   // 16B per lane per op
    __builtin_amdgcn_global_load_async_to_lds_b128(
        (g_v4i_ptr)(g + c * 4), (l_v4i_ptr)(l + c * 4), 0, 0);
#else
  for (int d = threadIdx.x; d < STEP_DW; d += TPB) l[d] = g[d];
#endif
}

__device__ __forceinline__ void wait_async() {
#if HAVE_ASYNC
#if __has_builtin(__builtin_amdgcn_s_wait_asynccnt)
  __builtin_amdgcn_s_wait_asynccnt(0);
#else
  asm volatile("s_wait_asynccnt 0" ::: "memory");
#endif
#endif
}

// Device-scope split barrier over NWG persistent workgroups (monotonic counter).
__device__ __forceinline__ void gsync(unsigned* bar, unsigned idx) {
  __syncthreads();
  if (threadIdx.x == 0) {
    __hip_atomic_fetch_add(bar, 1u, __ATOMIC_ACQ_REL, __HIP_MEMORY_SCOPE_AGENT);
    const unsigned target = idx * (unsigned)NWG;
    while (__hip_atomic_load(bar, __ATOMIC_ACQUIRE, __HIP_MEMORY_SCOPE_AGENT) < target)
      __builtin_amdgcn_s_sleep(1);
  }
  __syncthreads();
}

__global__ void init_h_kernel(const float* __restrict__ hidden,
                              uint16_t* __restrict__ h_buf) {
  int i = blockIdx.x * blockDim.x + threadIdx.x;
  if (i < BATCH * Hh) {
    int m = i >> 8, n = i & 255;
    h_buf[sw_u16(m, n)] = f2bf(hidden[i]);
  }
}

// One-time: embeddings f32 -> bf16, fragment-ordered per timestep (32 KB each).
__global__ void conv_x_kernel(const float* __restrict__ emb,
                              uint32_t* __restrict__ xsw) {
  const int t = blockIdx.x;
  for (int d = threadIdx.x; d < STEP_DW; d += blockDim.x) {
    int v = d & 7, ln = (d >> 3) & 31, kt8 = (d >> 8) & 7, mtt = d >> 11;
    int m = mtt * 16 + (ln & 15);
    int i = kt8 * 32 + (v >> 2) * 16 + (ln >> 4) * 8 + (v & 3) * 2;
    const float* e = emb + ((size_t)m * SEQ + t) * Hh + i;
    xsw[(size_t)t * STEP_DW + d] =
        (uint32_t)f2bf(e[0]) | ((uint32_t)f2bf(e[1]) << 16);
  }
}

__global__ __launch_bounds__(TPB, 1) void gru_persistent(
    const float* __restrict__ emb,
    const float* __restrict__ W_r, const float* __restrict__ b_r,
    const float* __restrict__ W_z, const float* __restrict__ b_z,
    const float* __restrict__ W_h, const float* __restrict__ b_h,
    float* __restrict__ out,
    unsigned* __restrict__ bar,
    uint16_t* __restrict__ h_buf,    // (64,256) bf16, fragment-ordered
    uint16_t* __restrict__ rh_buf,   // (64,256) bf16, fragment-ordered
    const uint32_t* __restrict__ xsw, int use_xsw) {

  __shared__ __align__(16) uint32_t AswH[4][8][32][8];     // h | r*h half, 32 KB
  __shared__ __align__(16) uint32_t AswX[2][4][8][32][8];  // x double buffer, 64 KB
  __shared__ __align__(16) uint32_t BswU[3][16][32][8];    // weights, 48 KB
  __shared__ float zs[BATCH][16];
  __shared__ float hs[BATCH][16];

  const int g     = blockIdx.x;
  const int nbase = g * 16;
  const int lane  = threadIdx.x & 31;
  const int mt    = threadIdx.x >> 5;            // wave -> M-tile
  const int nloc  = lane & 15;
  const int ncol  = nbase + nloc;
  const int mrow  = mt * 16 + (lane >> 4) * 8;   // C layout: m = mrow + j

  // ---- One-time: weights f32 -> bf16, WMMA-B swizzled, LDS-resident ----
  const float* Wg[3] = {W_r, W_z, W_h};
#pragma unroll
  for (int gate = 0; gate < 3; ++gate) {
    const float* W = Wg[gate];
    for (int d = threadIdx.x; d < 16 * 32 * 8; d += TPB) {
      int v = d & 7, ln = (d >> 3) & 31, kt = d >> 8;
      int nn = nbase + (ln & 15);
      int k  = kt * 32 + (v >> 2) * 16 + (ln >> 4) * 8 + (v & 3) * 2;
      uint32_t lo = f2bf(W[(size_t)k * Hh + nn]);
      uint32_t hi = f2bf(W[(size_t)(k + 1) * Hh + nn]);
      BswU[gate][kt][ln][v] = lo | (hi << 16);
    }
  }

  const float brv = b_r[ncol], bzv = b_z[ncol], bhv = b_h[ncol];
  unsigned bcnt = 0;

  if (use_xsw)   // prefetch x(0)
    stage_issue(xsw, &AswX[0][0][0][0][0]);

  for (int t = 0; t < SEQ; ++t) {
    // ---- Stage h (fragment-ordered 32 KB copy) + x tile ----
    stage_issue((const uint32_t*)h_buf, &AswH[0][0][0][0]);
    if (!use_xsw) {  // convert x_t on the fly into this step's buffer
      for (int d = threadIdx.x; d < STEP_DW; d += TPB) {
        int v = d & 7, ln = (d >> 3) & 31, kt8 = (d >> 8) & 7, mtt = d >> 11;
        int m = mtt * 16 + (ln & 15);
        int i = kt8 * 32 + (v >> 2) * 16 + (ln >> 4) * 8 + (v & 3) * 2;
        const float* e = emb + ((size_t)m * SEQ + t) * Hh + i;
        AswX[t & 1][mtt][kt8][ln][v] =
            (uint32_t)f2bf(e[0]) | ((uint32_t)f2bf(e[1]) << 16);
        if (v == 0 && t + 1 < SEQ)
          __builtin_prefetch(emb + ((size_t)m * SEQ + t + 1) * Hh + i, 0, 1);
      }
    }
    wait_async();
    __syncthreads();
    // overlap next step's x copy with Phase A and the first barrier
    if (use_xsw && t + 1 < SEQ)
      stage_issue(xsw + (size_t)(t + 1) * STEP_DW,
                  &AswX[(t + 1) & 1][0][0][0][0]);

    const uint32_t (*AX)[8][32][8] = AswX[t & 1];

    // ---- Phase A: [h|x] @ [Wr ; Wz] for this WG's 16 columns ----
    v8f ar = {}; v8f az = {};
#pragma unroll 4
    for (int kt = 0; kt < 8; ++kt) {
      v16bf a   = load_frag(&AswH[mt][kt][lane][0]);
      v16bf bfr = load_frag(&BswU[0][kt][lane][0]);
      v16bf bfz = load_frag(&BswU[1][kt][lane][0]);
      ar = __builtin_amdgcn_wmma_f32_16x16x32_bf16(false, a, false, bfr,
                                                   (short)0, ar, false, false);
      az = __builtin_amdgcn_wmma_f32_16x16x32_bf16(false, a, false, bfz,
                                                   (short)0, az, false, false);
    }
#pragma unroll 4
    for (int kt = 0; kt < 8; ++kt) {
      v16bf a   = load_frag(&AX[mt][kt][lane][0]);
      v16bf bfr = load_frag(&BswU[0][kt + 8][lane][0]);
      v16bf bfz = load_frag(&BswU[1][kt + 8][lane][0]);
      ar = __builtin_amdgcn_wmma_f32_16x16x32_bf16(false, a, false, bfr,
                                                   (short)0, ar, false, false);
      az = __builtin_amdgcn_wmma_f32_16x16x32_bf16(false, a, false, bfz,
                                                   (short)0, az, false, false);
    }
    // Epilogue A: sigmoids; stash z and h_old; publish r*h (swizzled) to L2.
    const uint16_t* AH16 = (const uint16_t*)AswH;
#pragma unroll
    for (int j = 0; j < 8; ++j) {
      int m   = mrow + j;
      float r = 1.0f / (1.0f + __expf(-(ar[j] + brv)));
      float z = 1.0f / (1.0f + __expf(-(az[j] + bzv)));
      float ho = bf2f(AH16[sw_u16(m, ncol)]);   // h_old from LDS stage
      zs[m][nloc] = z;
      hs[m][nloc] = ho;
      rh_buf[sw_u16(m, ncol)] = f2bf(r * ho);
    }
    __threadfence();
    gsync(bar, ++bcnt);

    // ---- Re-stage K rows 0..255 with full (r*h) from all WGs ----
    stage_issue((const uint32_t*)rh_buf, &AswH[0][0][0][0]);
    wait_async();
    __syncthreads();

    // ---- Phase B: [r*h|x] @ Wh ----
    v8f ah = {};
#pragma unroll 4
    for (int kt = 0; kt < 8; ++kt) {
      v16bf a   = load_frag(&AswH[mt][kt][lane][0]);
      v16bf bfh = load_frag(&BswU[2][kt][lane][0]);
      ah = __builtin_amdgcn_wmma_f32_16x16x32_bf16(false, a, false, bfh,
                                                   (short)0, ah, false, false);
    }
#pragma unroll 4
    for (int kt = 0; kt < 8; ++kt) {
      v16bf a   = load_frag(&AX[mt][kt][lane][0]);
      v16bf bfh = load_frag(&BswU[2][kt + 8][lane][0]);
      ah = __builtin_amdgcn_wmma_f32_16x16x32_bf16(false, a, false, bfh,
                                                   (short)0, ah, false, false);
    }
    // Epilogue B: blend, emit output row, publish new hidden (swizzled).
#pragma unroll
    for (int j = 0; j < 8; ++j) {
      int m    = mrow + j;
      float ht = tanhf(ah[j] + bhv);
      float z  = zs[m][nloc];
      float ho = hs[m][nloc];
      float hn = (1.0f - z) * ho + z * ht;
      out[((size_t)m * SEQ + t) * Hh + ncol] = hn;
      h_buf[sw_u16(m, ncol)] = f2bf(hn);
      if (t == SEQ - 1)
        out[(size_t)BATCH * SEQ * Hh + (size_t)m * Hh + ncol] = hn;
    }
    __threadfence();
    gsync(bar, ++bcnt);
  }
}

extern "C" void kernel_launch(void* const* d_in, const int* in_sizes, int n_in,
                              void* d_out, int out_size, void* d_ws,
                              size_t ws_size, hipStream_t stream) {
  const float* emb    = (const float*)d_in[0];
  const float* hidden = (const float*)d_in[1];
  const float* W_r    = (const float*)d_in[2];
  const float* b_r    = (const float*)d_in[3];
  const float* W_z    = (const float*)d_in[4];
  const float* b_z    = (const float*)d_in[5];
  const float* W_h    = (const float*)d_in[6];
  const float* b_h    = (const float*)d_in[7];
  float* out = (float*)d_out;

  // ws layout: [0,256) barrier; h 32 KB @256; r*h 32 KB @33024; xsw @65792.
  unsigned* bar    = (unsigned*)d_ws;
  uint16_t* h_buf  = (uint16_t*)((char*)d_ws + 256);
  uint16_t* rh_buf = (uint16_t*)((char*)d_ws + 33024);
  const size_t XSW_OFF   = 65792;
  const size_t XSW_BYTES = (size_t)SEQ * STEP_DW * 4;   // 64 MB
  uint32_t* xsw   = (uint32_t*)((char*)d_ws + XSW_OFF);
  int use_xsw     = (ws_size >= XSW_OFF + XSW_BYTES) ? 1 : 0;

  (void)hipMemsetAsync(d_ws, 0, 256, stream);
  init_h_kernel<<<(BATCH * Hh + 255) / 256, 256, 0, stream>>>(hidden, h_buf);
  if (use_xsw) conv_x_kernel<<<SEQ, 256, 0, stream>>>(emb, xsw);
  gru_persistent<<<NWG, TPB, 0, stream>>>(emb, W_r, b_r, W_z, b_z, W_h, b_h,
                                          out, bar, h_buf, rh_buf, xsw, use_xsw);
}
